// MultiHeadSelfAttention_1589137899896
// MI455X (gfx1250) — compile-verified, matrix operand order checked by reading, not run on hardware
//
#include <hip/hip_runtime.h>

typedef __attribute__((ext_vector_type(16))) _Float16 v16h;
typedef __attribute__((ext_vector_type(8)))  _Float16 v8h;
typedef __attribute__((ext_vector_type(8)))  float    v8f;
typedef int v4i_vs __attribute__((vector_size(16)));   // matches builtin's "int __vector(4)"

#define SEQ 4096
#define DM  1024
#define NH  16
#define DK  64

#define BSTRIDE 40   // padded LDS row stride in f16 (80B) -> conflict-free frag reads

// ---------- fragment helpers ----------

// Load 16 f16 for this lane: 8 contiguous at p, 8 contiguous at p+16 (elements).
// Matches documented 16-bit A-matrix packing (B assumed mirrored, lane = column).
__device__ __forceinline__ v16h load_frag(const _Float16* p) {
    v8h lo = *(const v8h*)p;
    v8h hi = *(const v8h*)(p + 16);
    v16h r;
#pragma unroll
    for (int i = 0; i < 8; ++i) { r[i] = lo[i]; r[i + 8] = hi[i]; }
    return r;
}

__device__ __forceinline__ v8f wmma_f16(v16h a, v16h b, v8f c) {
    return __builtin_amdgcn_wmma_f32_16x16x32_f16(
        /*neg_a=*/false, a, /*neg_b=*/false, b,
        /*c_mod=*/(short)0, c, /*reuse_a=*/false, /*reuse_b=*/false);
}

// ---------- async global -> LDS copy (16B per lane), feature-detected ----------

__device__ __forceinline__ void cp_b128_g2l(const _Float16* g, _Float16* l) {
#if __has_builtin(__builtin_amdgcn_global_load_async_to_lds_b128)
    __builtin_amdgcn_global_load_async_to_lds_b128(
        (__attribute__((address_space(1))) v4i_vs*)g,
        (__attribute__((address_space(3))) v4i_vs*)l,
        /*offset=*/0, /*cpol=*/0);
#else
    *(v8h*)l = *(const v8h*)g;   // synchronous fallback; __syncthreads() fences it
#endif
}

template <int N>
__device__ __forceinline__ void wait_async() {
#if __has_builtin(__builtin_amdgcn_s_wait_asynccnt)
    __builtin_amdgcn_s_wait_asynccnt(N);
#else
    if constexpr (N == 0) asm volatile("s_wait_asynccnt 0x0" ::: "memory");
    else                  asm volatile("s_wait_asynccnt 0x2" ::: "memory");
#endif
}

// ---------- f32 -> f16 conversion ----------

__global__ __launch_bounds__(256) void cvt_f32_f16(const float* __restrict__ in,
                                                   _Float16* __restrict__ out, int n) {
    int i = blockIdx.x * 256 + threadIdx.x;
    if (i < n) out[i] = (_Float16)in[i];
}

// ---------- GEMM: C[M,N] = A[M,K] * W[N,K]^T ----------
// grid = (N/128, M/128), block = 256 (8 waves as 4M x 2N).
// Wave tile 32x64. B block-tile (128 cols x 32 k) staged in LDS via async
// copies, double buffered; A fragments demand-loaded (unique per wave row).

template <bool OUTF32, bool TRANSP>
__global__ __launch_bounds__(256) void gemm16(const _Float16* __restrict__ A,
                                              const _Float16* __restrict__ W,
                                              void* __restrict__ C,
                                              int M, int N, int K) {
    __shared__ __align__(16) _Float16 Blds[2][128 * BSTRIDE];

    const int tid  = threadIdx.x;
    const int lane = tid & 31;
    const int wvid = tid >> 5;
    const int wm   = wvid & 3;          // wave row group 0..3
    const int wn   = wvid >> 2;         // wave col group 0..1
    const int m0   = blockIdx.y * 128 + wm * 32;
    const int nblk = blockIdx.x * 128;
    const int r    = lane & 15;
    const int kh   = lane >> 4;

    // cooperative staging: thread copies rows (tid>>2) and (tid>>2)+64, 8-f16 chunk (tid&3)
    const int srow = tid >> 2;
    const int scol = (tid & 3) * 8;
    const _Float16* wst = W + (size_t)(nblk + srow) * K + scol;

    v8f acc[2][4] = {};
    const _Float16* ap0 = A + (size_t)(m0 + r) * K + kh * 8;
    const _Float16* ap1 = A + (size_t)(m0 + 16 + r) * K + kh * 8;

    auto stage = [&](int buf, int k0) {
        cp_b128_g2l(wst + k0,                   &Blds[buf][srow * BSTRIDE + scol]);
        cp_b128_g2l(wst + (size_t)64 * K + k0,  &Blds[buf][(srow + 64) * BSTRIDE + scol]);
    };

    stage(0, 0);
    if (32 < K) stage(1, 32);

    for (int k0 = 0, i = 0; k0 < K; k0 += 32, ++i) {
        const int buf = i & 1;
        if (k0 + 32 < K) wait_async<2>();   // batch i done, batch i+1 may fly
        else             wait_async<0>();
        __syncthreads();

        v16h a0 = load_frag(ap0 + k0);
        v16h a1 = load_frag(ap1 + k0);
#pragma unroll
        for (int j = 0; j < 4; ++j) {
            const _Float16* bp = &Blds[buf][(wn * 64 + j * 16 + r) * BSTRIDE + kh * 8];
            v16h b = load_frag(bp);
            acc[0][j] = wmma_f16(a0, b, acc[0][j]);
            acc[1][j] = wmma_f16(a1, b, acc[1][j]);
        }
        __syncthreads();                    // all waves done reading buf
        if (k0 + 64 < K) stage(buf, k0 + 64);
    }

    const int n0 = nblk + wn * 64;
#pragma unroll
    for (int t = 0; t < 2; ++t) {
#pragma unroll
        for (int j = 0; j < 4; ++j) {
#pragma unroll
            for (int e = 0; e < 8; ++e) {
                int row = m0 + t * 16 + e + 8 * kh;
                int col = n0 + j * 16 + r;
                size_t idx = TRANSP ? (size_t)col * M + row : (size_t)row * N + col;
                float v = acc[t][j][e];
                if (OUTF32) ((float*)C)[idx] = v;
                else        ((_Float16*)C)[idx] = (_Float16)v;
            }
        }
    }
}

// ---------- RoPE on Q and K in place (f16) ----------

__global__ __launch_bounds__(256) void rope_qk(_Float16* __restrict__ q,
                                               _Float16* __restrict__ k) {
    int idx = blockIdx.x * 256 + threadIdx.x;       // SEQ * 512 pairs
    int s   = idx >> 9;
    int rem = idx & 511;
    int h   = rem >> 5;
    int j   = rem & 31;
    size_t base = (size_t)s * DM + h * DK + 2 * j;

    float inv = __powf(10000.0f, -(float)j * (1.0f / 32.0f));
    float ang = (float)s * inv;
    float sn, cs;
    __sincosf(ang, &sn, &cs);

    float qe = (float)q[base], qo = (float)q[base + 1];
    q[base]     = (_Float16)(cs * qe - sn * qo);
    q[base + 1] = (_Float16)(sn * qe + cs * qo);

    float ke = (float)k[base], ko = (float)k[base + 1];
    k[base]     = (_Float16)(cs * ke - sn * ko);
    k[base + 1] = (_Float16)(sn * ke + cs * ko);
}

// ---------- Flash attention: causal softmax(Q K^T / 8) V ----------
// grid = (SEQ/128, NH), block = 256 (8 waves); each wave owns one 16-query tile.
// Q,K layout: [s][h*64+d] (ld=DM).  V layout: transposed Vt[h][d][s] (ld=SEQ).

__global__ __launch_bounds__(256) void attn_fwd(const _Float16* __restrict__ Q,
                                                const _Float16* __restrict__ K,
                                                const _Float16* __restrict__ Vt,
                                                _Float16* __restrict__ O) {
    __shared__ __align__(16) _Float16 plds[8 * 16 * 32];   // per-wave 16x32 P tile

    const int lane = threadIdx.x & 31;
    const int wv   = threadIdx.x >> 5;
    const int h    = blockIdx.y;
    const int q0   = blockIdx.x * 128 + wv * 16;
    const int r    = lane & 15;
    const int kh   = lane >> 4;

    const _Float16* Qh = Q + h * DK;
    const _Float16* Kh = K + h * DK;
    const _Float16* Vh = Vt + (size_t)h * DK * SEQ;
    _Float16* lp = plds + wv * (16 * 32);

    const _Float16* qp = Qh + (size_t)(q0 + r) * DM + kh * 8;
    v16h aq0 = load_frag(qp);
    v16h aq1 = load_frag(qp + 32);

    float m[8], l[8];
    v8f acc[4] = {};
#pragma unroll
    for (int e = 0; e < 8; ++e) { m[e] = -1e30f; l[e] = 0.0f; }

    const float scale = 0.125f;   // 1/sqrt(64)

    for (int kb = 0; kb < q0 + 16; kb += 32) {
        const _Float16* kp = Kh + (size_t)(kb + r) * DM + kh * 8;

        // prefetch next chunk's K rows and V rows into cache
        if (kb + 32 < q0 + 16) {
            __builtin_prefetch(kp + 32 * DM, 0, 1);
            __builtin_prefetch(kp + 48 * DM, 0, 1);
            __builtin_prefetch(Vh + (size_t)r * SEQ + kb + 32, 0, 1);
        }

        v16h bk00 = load_frag(kp);                 // keys kb..+15,   d 0..31
        v16h bk01 = load_frag(kp + 32);            // keys kb..+15,   d 32..63
        v16h bk10 = load_frag(kp + 16 * DM);       // keys kb+16..+31, d 0..31
        v16h bk11 = load_frag(kp + 16 * DM + 32);  // keys kb+16..+31, d 32..63

        v8f s0 = {}, s1 = {};
        s0 = wmma_f16(aq0, bk00, s0);
        s0 = wmma_f16(aq1, bk01, s0);
        s1 = wmma_f16(aq0, bk10, s1);
        s1 = wmma_f16(aq1, bk11, s1);

        const int kc0 = kb + r, kc1 = kb + 16 + r;
#pragma unroll
        for (int e = 0; e < 8; ++e) {
            int qrow = q0 + e + 8 * kh;           // C-layout row mapping
            float x0 = (kc0 <= qrow) ? s0[e] * scale : -1e30f;
            float x1 = (kc1 <= qrow) ? s1[e] * scale : -1e30f;

            float v = fmaxf(x0, x1);              // 16-lane half-wave row max
            v = fmaxf(v, __shfl_xor(v, 1, 32));
            v = fmaxf(v, __shfl_xor(v, 2, 32));
            v = fmaxf(v, __shfl_xor(v, 4, 32));
            v = fmaxf(v, __shfl_xor(v, 8, 32));
            float mn = fmaxf(m[e], v);

            float alpha = __expf(m[e] - mn);
            float e0 = __expf(x0 - mn);
            float e1 = __expf(x1 - mn);
            float sum = e0 + e1;
            sum += __shfl_xor(sum, 1, 32);
            sum += __shfl_xor(sum, 2, 32);
            sum += __shfl_xor(sum, 4, 32);
            sum += __shfl_xor(sum, 8, 32);

            l[e] = l[e] * alpha + sum;
            m[e] = mn;
#pragma unroll
            for (int j = 0; j < 4; ++j) acc[j][e] *= alpha;

            lp[(e + 8 * kh) * 32 + r]      = (_Float16)e0;
            lp[(e + 8 * kh) * 32 + 16 + r] = (_Float16)e1;
        }

        asm volatile("s_wait_dscnt 0" ::: "memory");   // wave-local LDS ordering

        v16h pa = load_frag(lp + r * 32 + kh * 8);     // P as A-fragment

#pragma unroll
        for (int j = 0; j < 4; ++j) {
            v16h bv = load_frag(Vh + (size_t)(j * 16 + r) * SEQ + kb + kh * 8);
            acc[j] = wmma_f16(pa, bv, acc[j]);
        }
    }

    _Float16* op = O + h * DK;
#pragma unroll
    for (int j = 0; j < 4; ++j) {
#pragma unroll
        for (int e = 0; e < 8; ++e) {
            int row = q0 + e + 8 * kh;
            op[(size_t)row * DM + j * 16 + r] = (_Float16)(acc[j][e] / l[e]);
        }
    }
}

// ---------- launch ----------

extern "C" void kernel_launch(void* const* d_in, const int* in_sizes, int n_in,
                              void* d_out, int out_size, void* d_ws, size_t ws_size,
                              hipStream_t stream) {
    (void)in_sizes; (void)n_in; (void)out_size; (void)ws_size;

    const float* x  = (const float*)d_in[0];
    const float* wq = (const float*)d_in[1];
    const float* wk = (const float*)d_in[2];
    const float* wv = (const float*)d_in[3];
    const float* wo = (const float*)d_in[4];
    float* out = (float*)d_out;

    const size_t NX = (size_t)SEQ * DM;   // 4M elems
    const size_t NW = (size_t)DM * DM;    // 1M elems

    _Float16* p = (_Float16*)d_ws;
    _Float16* xh  = p;  p += NX;
    _Float16* wqh = p;  p += NW;
    _Float16* wkh = p;  p += NW;
    _Float16* wvh = p;  p += NW;
    _Float16* woh = p;  p += NW;
    _Float16* qh  = p;  p += NX;
    _Float16* kh  = p;  p += NX;
    _Float16* vth = p;  p += NX;          // transposed V: [h][d][s]
    _Float16* ah  = p;  p += NX;          // attention output (pre-proj)

    // 0) f32 -> f16 conversions
    cvt_f32_f16<<<(int)((NX + 255) / 256), 256, 0, stream>>>(x,  xh,  (int)NX);
    cvt_f32_f16<<<(int)((NW + 255) / 256), 256, 0, stream>>>(wq, wqh, (int)NW);
    cvt_f32_f16<<<(int)((NW + 255) / 256), 256, 0, stream>>>(wk, wkh, (int)NW);
    cvt_f32_f16<<<(int)((NW + 255) / 256), 256, 0, stream>>>(wv, wvh, (int)NW);
    cvt_f32_f16<<<(int)((NW + 255) / 256), 256, 0, stream>>>(wo, woh, (int)NW);

    // 1) Q, K, V projections (V stored transposed for contiguous PV B-frags)
    dim3 gg(DM / 128, SEQ / 128);
    gemm16<false, false><<<gg, 256, 0, stream>>>(xh, wqh, qh,  SEQ, DM, DM);
    gemm16<false, false><<<gg, 256, 0, stream>>>(xh, wkh, kh,  SEQ, DM, DM);
    gemm16<false, true ><<<gg, 256, 0, stream>>>(xh, wvh, vth, SEQ, DM, DM);

    // 2) RoPE on Q and K
    rope_qk<<<(SEQ * (DM / 2)) / 256, 256, 0, stream>>>(qh, kh);

    // 3) causal flash attention
    dim3 ga(SEQ / 128, NH);
    attn_fwd<<<ga, 256, 0, stream>>>(qh, kh, vth, ah);

    // 4) output projection, f32 result straight to d_out
    gemm16<true, false><<<gg, 256, 0, stream>>>(ah, woh, out, SEQ, DM, DM);
}
